// cross_Attention_38620345926297
// MI455X (gfx1250) — matrix operand order repaired and checked by reading, not compiled, not run on hardware
//
#include <hip/hip_runtime.h>
#include <hip/hip_bf16.h>
#include <math.h>

#define BB  8
#define TT  512
#define CC  1024
#define NHH 16
#define DHH 64
#define HWW 256

typedef __attribute__((ext_vector_type(16))) __bf16 v16bf;
typedef __attribute__((ext_vector_type(8)))  __bf16 v8bf;
typedef __attribute__((ext_vector_type(8)))  float  v8f;

// ---------------------------------------------------------------------------
// Fragment loaders from LDS, per CDNA5 ISA layouts (CDNA5 ISA):
// A 16x32 bf16 : lane<16  -> row=lane,    K {0..7} in v0..3,  K {16..23} in v4..7
//                lane>=16 -> row=lane-16, K {8..15},          K {24..31}
// B 32x16 bf16 : lane<16  -> col=lane,    K {0..7} in v0..3,  K {16..23} in v4..7
//                lane>=16 -> col=lane-16, K {8..15},          K {24..31}
// tile pointers must be 16-byte aligned, stride (in elements) multiple of 8.
// ---------------------------------------------------------------------------
__device__ __forceinline__ v16bf frag_a_ld(const __bf16* p0, int stride, int lane) {
  const int m  = lane & 15;
  const int k0 = (lane >> 4) << 3;            // 0 or 8
  const __bf16* p = p0 + m * stride + k0;
  v8bf lo = *(const v8bf*)(p);
  v8bf hi = *(const v8bf*)(p + 16);
  v16bf a;
#pragma unroll
  for (int i = 0; i < 8; ++i) { a[i] = lo[i]; a[i + 8] = hi[i]; }
  return a;
}

__device__ __forceinline__ v16bf frag_b_ld(const __bf16* p0, int stride, int lane) {
  const int n  = lane & 15;
  const int k0 = (lane >> 4) << 3;            // 0 or 8
  const __bf16* p = p0 + n * stride + k0;
  v8bf lo = *(const v8bf*)(p);
  v8bf hi = *(const v8bf*)(p + 16);
  v16bf b;
#pragma unroll
  for (int i = 0; i < 8; ++i) { b[i] = lo[i]; b[i + 8] = hi[i]; }
  return b;
}

__device__ __forceinline__ v8f wmma_bf16(v16bf a, v16bf b, v8f c) {
  return __builtin_amdgcn_wmma_f32_16x16x32_bf16(false, a, false, b,
                                                 (short)0, c, false, false);
}

// ---------------------------------------------------------------------------
// CDNA5 async memory->LDS tile fill (GLOBAL_LOAD_ASYNC_TO_LDS_B128, ASYNCcnt).
// Copies a 64-row x 128-byte tile; LDS row stride 144 B (72 bf16, 16B-aligned).
// GVS addressing: uniform SGPR base + per-lane 32-bit byte offset.
// LDS destination address = low 32 bits of the flat pointer.
// ---------------------------------------------------------------------------
__device__ __forceinline__ void async_tile_ld(const __bf16* gbase,
                                              unsigned lds_base,
                                              int grow_stride_bytes, int tid) {
#pragma unroll
  for (int i = 0; i < 4; ++i) {
    int id = tid + i * 128;
    int r = id >> 3, c = id & 7;
    unsigned lds  = lds_base + r * 144 + c * 16;
    unsigned goff = (unsigned)(r * grow_stride_bytes + c * 16);
    asm volatile("global_load_async_to_lds_b128 %0, %1, %2"
                 :: "v"(lds), "v"(goff), "s"(gbase) : "memory");
  }
}

__device__ __forceinline__ void wait_async0() {
  asm volatile("s_wait_asynccnt 0x0" ::: "memory");
}

// ---------------------------------------------------------------------------
// Kernel 1: fused QKV projection.  y = inp @ W^T + bias  (NT GEMM, fp32->bf16)
// blockIdx.z: 0 = Q(x), 1 = K(src), 2 = V(src, stored transposed [b,h,d,t])
// block = 128 threads (4 waves), tile = 64(M) x 64(N), K-step 32.
// fp32->bf16 conversion happens at LDS staging (so no async path here).
// ---------------------------------------------------------------------------
__global__ __launch_bounds__(128)
void qkv_proj_kernel(const float* __restrict__ x, const float* __restrict__ src,
                     const float* __restrict__ Wq, const float* __restrict__ bq,
                     const float* __restrict__ Wk, const float* __restrict__ bk,
                     const float* __restrict__ Wv, const float* __restrict__ bv,
                     __bf16* __restrict__ Qb, __bf16* __restrict__ Kb,
                     __bf16* __restrict__ Vt)
{
  __shared__ __attribute__((aligned(16))) __bf16 As[64][40];
  __shared__ __attribute__((aligned(16))) __bf16 Ws[64][40];
  const int tid  = threadIdx.x;
  const int wave = tid >> 5, lane = tid & 31, half = lane >> 4;
  const int nBase = blockIdx.x * 64;
  const int mBase = blockIdx.y * 64;
  const int which = blockIdx.z;

  const float* inp  = (which == 0) ? x  : src;
  const float* W    = (which == 0) ? Wq : (which == 1) ? Wk : Wv;
  const float* bias = (which == 0) ? bq : (which == 1) ? bk : bv;

  v8f acc[4] = {};
  for (int kk = 0; kk < CC; kk += 32) {
    for (int i = tid; i < 64 * 32; i += 128) {
      int r = i >> 5, c = i & 31;
      As[r][c] = (__bf16)inp[(size_t)(mBase + r) * CC + kk + c];
      Ws[r][c] = (__bf16)W  [(size_t)(nBase + r) * CC + kk + c];
    }
    __syncthreads();
    v16bf a = frag_a_ld(&As[wave * 16][0], 40, lane);
#pragma unroll
    for (int j = 0; j < 4; ++j) {
      v16bf bf = frag_b_ld(&Ws[j * 16][0], 40, lane);
      acc[j] = wmma_bf16(a, bf, acc[j]);
    }
    __syncthreads();
  }

#pragma unroll
  for (int j = 0; j < 4; ++j) {
#pragma unroll
    for (int r = 0; r < 8; ++r) {
      int row = mBase + wave * 16 + r + half * 8;
      int col = nBase + j * 16 + (lane & 15);
      float val = acc[j][r] + bias[col];
      if (which == 0) {
        Qb[(size_t)row * CC + col] = (__bf16)val;
      } else if (which == 1) {
        Kb[(size_t)row * CC + col] = (__bf16)val;
      } else {
        int bidx = row >> 9, t = row & 511;
        int hh = col >> 6,   d = col & 63;
        Vt[(((size_t)(bidx * NHH + hh)) * DHH + d) * TT + t] = (__bf16)val;
      }
    }
  }
}

// ---------------------------------------------------------------------------
// Kernel 2: flash attention per (q-tile=64, head, batch).
// S = Q K^T * 1/8, epipolar mask on [0:256,0:256], online softmax, O += P V.
// K/Vt tiles are double-buffered in LDS and filled with
// global_load_async_to_lds_b128 so the next tile streams in while WMMAs run.
// block = 128 threads (4 waves); each wave owns 16 q-rows.
// ---------------------------------------------------------------------------
__global__ __launch_bounds__(128)
void flash_attn_kernel(const __bf16* __restrict__ Qb, const __bf16* __restrict__ Kb,
                       const __bf16* __restrict__ Vt,
                       const float* __restrict__ fmap, const float* __restrict__ bmap,
                       __bf16* __restrict__ Y)
{
  __shared__ __attribute__((aligned(16))) __bf16 Qs[64][72];
  __shared__ __attribute__((aligned(16))) __bf16 Ks[2][64][72];
  __shared__ __attribute__((aligned(16))) __bf16 Vs[2][64][72];
  __shared__ __attribute__((aligned(16))) __bf16 Ps[64][72];
  const int tid  = threadIdx.x;
  const int wave = tid >> 5, lane = tid & 31, half = lane >> 4;
  const int qBase = blockIdx.x * 64;
  const int h = blockIdx.y;
  const int b = blockIdx.z;
  const float scale = 0.125f;  // 1/sqrt(64)

  const __bf16* kb_head = Kb + (size_t)(b * TT) * CC + h * DHH;           // +row*CC
  const __bf16* vt_head = Vt + ((size_t)(b * NHH + h)) * DHH * TT;        // +d*TT

  // async prefetch: Q tile + first K/V tiles
  async_tile_ld(Qb + (size_t)(b * TT + qBase) * CC + h * DHH,
                (unsigned)(uintptr_t)&Qs[0][0], CC * 2, tid);
  async_tile_ld(kb_head, (unsigned)(uintptr_t)&Ks[0][0][0], CC * 2, tid);
  async_tile_ld(vt_head, (unsigned)(uintptr_t)&Vs[0][0][0], TT * 2, tid);
  wait_async0();
  __syncthreads();

  v8f o[4] = {};
  float rmax[8], rsum[8];
#pragma unroll
  for (int r = 0; r < 8; ++r) { rmax[r] = -1e30f; rsum[r] = 0.f; }

  for (int kt = 0; kt < 8; ++kt) {
    const int kBase = kt * 64;
    const int buf = kt & 1;

    // stream next K/V tile into the alternate buffer while we compute
    if (kt < 7) {
      async_tile_ld(kb_head + (size_t)(kBase + 64) * CC,
                    (unsigned)(uintptr_t)&Ks[buf ^ 1][0][0], CC * 2, tid);
      async_tile_ld(vt_head + (kBase + 64),
                    (unsigned)(uintptr_t)&Vs[buf ^ 1][0][0], TT * 2, tid);
    }

    // S = Q K^T (NT: B-columns are K rows)
    v8f s[4] = {};
#pragma unroll
    for (int ks = 0; ks < 64; ks += 32) {
      v16bf a = frag_a_ld(&Qs[wave * 16][ks], 72, lane);
#pragma unroll
      for (int j = 0; j < 4; ++j) {
        v16bf bf = frag_b_ld(&Ks[buf][j * 16][ks], 72, lane);
        s[j] = wmma_bf16(a, bf, s[j]);
      }
    }

    // scale + bidirectional epipolar mask (multiplies logits, pre-softmax)
#pragma unroll
    for (int j = 0; j < 4; ++j) {
#pragma unroll
      for (int r = 0; r < 8; ++r) {
        int qg = qBase + wave * 16 + r + half * 8;
        int kg = kBase + j * 16 + (lane & 15);
        float sv = s[j][r] * scale;
        if (qg < HWW && kg < HWW) {
          sv *= bmap[((size_t)b * HWW + kg) * HWW + qg] *
                fmap[((size_t)b * HWW + qg) * HWW + kg];
        }
        s[j][r] = sv;
      }
    }

    // online softmax; D-layout rows live in 16-lane groups -> shfl_xor 1,2,4,8
#pragma unroll
    for (int r = 0; r < 8; ++r) {
      float m = fmaxf(fmaxf(s[0][r], s[1][r]), fmaxf(s[2][r], s[3][r]));
#pragma unroll
      for (int off = 1; off < 16; off <<= 1)
        m = fmaxf(m, __shfl_xor(m, off, 32));
      float nm   = fmaxf(rmax[r], m);
      float corr = __expf(rmax[r] - nm);
      rmax[r] = nm;
      float psum = 0.f;
#pragma unroll
      for (int j = 0; j < 4; ++j) {
        float p = __expf(s[j][r] - nm);
        s[j][r] = p;
        psum += p;
      }
#pragma unroll
      for (int off = 1; off < 16; off <<= 1)
        psum += __shfl_xor(psum, off, 32);
      rsum[r] = rsum[r] * corr + psum;
#pragma unroll
      for (int t = 0; t < 4; ++t) o[t][r] *= corr;
      // stage P (D-layout -> row-major LDS); wave-private rows, DS in-order
#pragma unroll
      for (int j = 0; j < 4; ++j)
        Ps[wave * 16 + r + half * 8][j * 16 + (lane & 15)] = (__bf16)s[j][r];
    }

    // O += P @ V  (B-columns = Vt rows, contiguous over k)
#pragma unroll
    for (int ks = 0; ks < 64; ks += 32) {
      v16bf pa = frag_a_ld(&Ps[wave * 16][ks], 72, lane);
#pragma unroll
      for (int t = 0; t < 4; ++t) {
        v16bf vb = frag_b_ld(&Vs[buf][t * 16][ks], 72, lane);
        o[t] = wmma_bf16(pa, vb, o[t]);
      }
    }

    if (kt < 7) wait_async0();  // next tile landed in LDS (per-wave counter)
    __syncthreads();            // publish to all waves; buf safe to overwrite
  }

  // normalize and write merged-head Y (bf16)
#pragma unroll
  for (int t = 0; t < 4; ++t) {
#pragma unroll
    for (int r = 0; r < 8; ++r) {
      int row = b * TT + qBase + wave * 16 + r + half * 8;
      int col = h * DHH + t * 16 + (lane & 15);
      Y[(size_t)row * CC + col] = (__bf16)(o[t][r] / rsum[r]);
    }
  }
}

// ---------------------------------------------------------------------------
// Kernel 3: output projection.  out = Y @ Wp^T + bp   (fp32 out)
// ---------------------------------------------------------------------------
__global__ __launch_bounds__(128)
void out_proj_kernel(const __bf16* __restrict__ Y, const float* __restrict__ Wp,
                     const float* __restrict__ bp, float* __restrict__ out)
{
  __shared__ __attribute__((aligned(16))) __bf16 As[64][40];
  __shared__ __attribute__((aligned(16))) __bf16 Ws[64][40];
  const int tid  = threadIdx.x;
  const int wave = tid >> 5, lane = tid & 31, half = lane >> 4;
  const int nBase = blockIdx.x * 64;
  const int mBase = blockIdx.y * 64;

  v8f acc[4] = {};
  for (int kk = 0; kk < CC; kk += 32) {
    for (int i = tid; i < 64 * 32; i += 128) {
      int r = i >> 5, c = i & 31;
      As[r][c] = Y[(size_t)(mBase + r) * CC + kk + c];
      Ws[r][c] = (__bf16)Wp[(size_t)(nBase + r) * CC + kk + c];
    }
    __syncthreads();
    v16bf a = frag_a_ld(&As[wave * 16][0], 40, lane);
#pragma unroll
    for (int j = 0; j < 4; ++j) {
      v16bf bf = frag_b_ld(&Ws[j * 16][0], 40, lane);
      acc[j] = wmma_bf16(a, bf, acc[j]);
    }
    __syncthreads();
  }

#pragma unroll
  for (int j = 0; j < 4; ++j) {
#pragma unroll
    for (int r = 0; r < 8; ++r) {
      int row = mBase + wave * 16 + r + half * 8;
      int col = nBase + j * 16 + (lane & 15);
      out[(size_t)row * CC + col] = acc[j][r] + bp[col];
    }
  }
}

// ---------------------------------------------------------------------------
extern "C" void kernel_launch(void* const* d_in, const int* in_sizes, int n_in,
                              void* d_out, int out_size, void* d_ws, size_t ws_size,
                              hipStream_t stream) {
  const float* x    = (const float*)d_in[0];
  const float* src  = (const float*)d_in[1];
  const float* fmap = (const float*)d_in[2];
  const float* bmap = (const float*)d_in[3];
  const float* Wq   = (const float*)d_in[4];
  const float* bq   = (const float*)d_in[5];
  const float* Wk   = (const float*)d_in[6];
  const float* bk   = (const float*)d_in[7];
  const float* Wv   = (const float*)d_in[8];
  const float* bv   = (const float*)d_in[9];
  const float* Wp   = (const float*)d_in[10];
  const float* bp   = (const float*)d_in[11];

  const size_t N = (size_t)BB * TT * CC;  // 4M elements
  __bf16* Qb = (__bf16*)d_ws;
  __bf16* Kb = Qb + N;
  __bf16* Vt = Qb + 2 * N;
  __bf16* Yb = Qb + 3 * N;

  qkv_proj_kernel<<<dim3(CC / 64, (BB * TT) / 64, 3), 128, 0, stream>>>(
      x, src, Wq, bq, Wk, bk, Wv, bv, Qb, Kb, Vt);
  flash_attn_kernel<<<dim3(TT / 64, NHH, BB), 128, 0, stream>>>(
      Qb, Kb, Vt, fmap, bmap, Yb);
  out_proj_kernel<<<dim3(CC / 64, (BB * TT) / 64), 128, 0, stream>>>(
      Yb, Wp, bp, (float*)d_out);
}
